// EvolutionaryClusterVQ_63780264346101
// MI455X (gfx1250) — compile-verified, matrix operand order checked by reading, not run on hardware
//
#include <hip/hip_runtime.h>

// ---------------------------------------------------------------------------
// EvolutionaryClusterVQ on MI455X (gfx1250, wave32, WMMA)
//
// inputs  : 16 x 4096 x 128 f32  -> 65536 rows x 128
// w_shape : 1024 x 64 f32 (unit rows) ; w_color : 16 x 64 f32 (unit rows)
// d_out (float32, concatenated): quantized (8388608) | vq_loss (1) |
//                                s_idx (65536) | c_idx (65536)
//
// Roofline: sim matmul = 65536x1024x64 MAC (~8.6 GFLOP) -> f16 WMMA with f32
// accumulate makes it ~free; HBM traffic ~66 MB (input read + output write,
// re-reads hit the 192 MB L2) => ~3 us memory-bound target.
// Codebook (f16, 130 KB) lives in LDS (CDNA5: 320 KB/WGP).
// Loss uses per-tile partials + fixed-order reduction => deterministic.
// Output stores are non-temporal so they don't evict the input from L2.
// ---------------------------------------------------------------------------

#define N_ROWS   65536
#define D        128
#define HALF_D   64
#define KS       1024          // shape codebook entries
#define KC       16            // color codebook entries
#define N_TILES  (N_ROWS / 16) // 4096 row-tiles of 16
#define EPSN     1e-12f

typedef __attribute__((ext_vector_type(16))) _Float16 v16h;
typedef __attribute__((ext_vector_type(8)))  _Float16 v8h;
typedef __attribute__((ext_vector_type(8)))  float    v8f;
typedef __attribute__((ext_vector_type(4)))  float    v4f;

static __device__ __forceinline__ v8f wmma_f16(v16h a, v16h b, v8f c) {
  // 8 args: (neg_a, A, neg_b, B, c_mod, C, reuse_a, reuse_b)
  return __builtin_amdgcn_wmma_f32_16x16x32_f16(false, a, false, b, (short)0, c,
                                                false, false);
}

// A fragment, 16-bit 16x32 layout (ISA 7.12.2):
//   lanes 0-15 : row M=lane,    VGPR0-3 -> K=0..7,  VGPR4-7 -> K=16..23
//   lanes16-31 : row M=lane-16, VGPR0-3 -> K=8..15, VGPR4-7 -> K=24..31
static __device__ __forceinline__ v16h load_a_frag(const float* rp, int ck,
                                                   int kb0, int kb1, float inv) {
  v4f q0 = *(const v4f*)(rp + ck + kb0);
  v4f q1 = *(const v4f*)(rp + ck + kb0 + 4);
  v4f q2 = *(const v4f*)(rp + ck + kb1);
  v4f q3 = *(const v4f*)(rp + ck + kb1 + 4);
  v16h a;
  a[0]  = (_Float16)(q0.x * inv); a[1]  = (_Float16)(q0.y * inv);
  a[2]  = (_Float16)(q0.z * inv); a[3]  = (_Float16)(q0.w * inv);
  a[4]  = (_Float16)(q1.x * inv); a[5]  = (_Float16)(q1.y * inv);
  a[6]  = (_Float16)(q1.z * inv); a[7]  = (_Float16)(q1.w * inv);
  a[8]  = (_Float16)(q2.x * inv); a[9]  = (_Float16)(q2.y * inv);
  a[10] = (_Float16)(q2.z * inv); a[11] = (_Float16)(q2.w * inv);
  a[12] = (_Float16)(q3.x * inv); a[13] = (_Float16)(q3.y * inv);
  a[14] = (_Float16)(q3.z * inv); a[15] = (_Float16)(q3.w * inv);
  return a;
}

// B fragment from LDS (code row = B column, same K striping as A).
static __device__ __forceinline__ v16h load_b_frag(const _Float16* crow, int ck,
                                                   int kb0, int kb1) {
  v8h lo = *(const v8h*)(crow + ck + kb0);   // ds_load_b128 (16B aligned)
  v8h hi = *(const v8h*)(crow + ck + kb1);
  return __builtin_shufflevector(lo, hi, 0, 1, 2, 3, 4, 5, 6, 7,
                                 8, 9, 10, 11, 12, 13, 14, 15);
}

// --------------------------------------------------------------------------
// Kernel 1: renormalize codebooks (mirrors _normalize(W)), convert to f16,
//           pack [shape | color] into workspace. One wave per code row.
// --------------------------------------------------------------------------
__global__ __launch_bounds__(256) void vq_prep(const float* __restrict__ wS,
                                               const float* __restrict__ wC,
                                               _Float16* __restrict__ cb) {
  int wid  = (blockIdx.x * blockDim.x + threadIdx.x) >> 5;  // global wave id
  int lane = threadIdx.x & 31;
  if (wid >= KS + KC) return;
  const float* src = (wid < KS) ? (wS + wid * HALF_D)
                                : (wC + (wid - KS) * HALF_D);
  float a0 = src[lane];
  float a1 = src[lane + 32];
  float s = a0 * a0 + a1 * a1;
  #pragma unroll
  for (int m = 1; m < 32; m <<= 1) s += __shfl_xor(s, m, 32);
  float inv = 1.0f / fmaxf(sqrtf(s), EPSN);
  _Float16* dst = cb + wid * HALF_D;
  dst[lane]      = (_Float16)(a0 * inv);
  dst[lane + 32] = (_Float16)(a1 * inv);
}

// --------------------------------------------------------------------------
// Kernel 2: main VQ. 8 waves/block, each wave owns 16-row tiles.
// --------------------------------------------------------------------------
__global__ __launch_bounds__(256, 1) void vq_main(
    const float* __restrict__ in, const float* __restrict__ wS,
    const float* __restrict__ wC, const _Float16* __restrict__ cb,
    float* __restrict__ outQ, float* __restrict__ outS,
    float* __restrict__ outC, float* __restrict__ lossPartial) {
  // Entire f16 codebook resident in LDS: (1024+16)*64 halves = 130 KB.
  __shared__ __align__(16) _Float16 sCB[(KS + KC) * HALF_D];

  {
    const uint4* gsrc = (const uint4*)cb;
    uint4* ldst = (uint4*)sCB;
    const int nvec = (KS + KC) * HALF_D * 2 / 16;  // 8320 x 16B
    for (int i = threadIdx.x; i < nvec; i += blockDim.x) ldst[i] = gsrc[i];
  }
  __syncthreads();

  const int lane  = threadIdx.x & 31;
  const int wave  = threadIdx.x >> 5;
  const int lhalf = lane & 15;              // column / row-in-tile id
  const int hi    = lane >> 4;              // which K sub-stripe
  const int kb0   = hi ? 8 : 0;
  const int kb1   = hi ? 24 : 16;
  const int wstride = gridDim.x * 8;

  for (int tile = blockIdx.x * 8 + wave; tile < N_TILES; tile += wstride) {
    const int rowBase = tile * 16;

    if (tile + wstride < N_TILES)  // speculative prefetch of next tile's rows
      __builtin_prefetch(in + (size_t)(tile + wstride) * 16 * D, 0, 1);

    // ---- per-row, per-half inverse norms: lane = hi*16 + row -------------
    const float* nr = in + (size_t)(rowBase + lhalf) * D + hi * HALF_D;
    float ss = 0.0f;
    #pragma unroll
    for (int j = 0; j < 16; ++j) {
      v4f v = ((const v4f*)nr)[j];
      ss += v.x * v.x + v.y * v.y + v.z * v.z + v.w * v.w;
    }
    float inv  = 1.0f / fmaxf(sqrtf(ss), EPSN);
    float invS = __shfl(inv, lhalf, 32);       // shape inv-norm for my row
    float invC = __shfl(inv, 16 + lhalf, 32);  // color inv-norm for my row

    // ---- A fragments (normalized rows, f16) ------------------------------
    const float* rp = in + (size_t)(rowBase + lhalf) * D;
    v16h aS0 = load_a_frag(rp, 0, kb0, kb1, invS);
    v16h aS1 = load_a_frag(rp, 32, kb0, kb1, invS);
    v16h aC0 = load_a_frag(rp + HALF_D, 0, kb0, kb1, invC);
    v16h aC1 = load_a_frag(rp + HALF_D, 32, kb0, kb1, invC);

    // ---- shape argmax over 1024 codes: 64 tiles x 2 WMMA -----------------
    float bv[8];
    int   bi[8];
    #pragma unroll
    for (int j = 0; j < 8; ++j) { bv[j] = -3.0e38f; bi[j] = 0; }
    #pragma unroll 4
    for (int t = 0; t < KS / 16; ++t) {
      const _Float16* crow = sCB + (size_t)(t * 16 + lhalf) * HALF_D;
      v16h b0 = load_b_frag(crow, 0, kb0, kb1);
      v16h b1 = load_b_frag(crow, 32, kb0, kb1);
      v8f acc = {};
      acc = wmma_f16(aS0, b0, acc);
      acc = wmma_f16(aS1, b1, acc);
      int col = t * 16 + lhalf;
      #pragma unroll
      for (int j = 0; j < 8; ++j)
        if (acc[j] > bv[j]) { bv[j] = acc[j]; bi[j] = col; }  // keep first max
    }
    // cross-lane reduce within 16-lane halves (C layout: slot j = row j / j+8)
    #pragma unroll
    for (int m = 1; m <= 8; m <<= 1) {
      #pragma unroll
      for (int j = 0; j < 8; ++j) {
        float ov = __shfl_xor(bv[j], m, 32);
        int   oi = __shfl_xor(bi[j], m, 32);
        if (ov > bv[j] || (ov == bv[j] && oi < bi[j])) { bv[j] = ov; bi[j] = oi; }
      }
    }

    // ---- color argmax over 16 codes: 1 tile x 2 WMMA ---------------------
    float cv[8];
    int   ci[8];
    {
      const _Float16* crow = sCB + (size_t)(KS + lhalf) * HALF_D;
      v16h b0 = load_b_frag(crow, 0, kb0, kb1);
      v16h b1 = load_b_frag(crow, 32, kb0, kb1);
      v8f acc = {};
      acc = wmma_f16(aC0, b0, acc);
      acc = wmma_f16(aC1, b1, acc);
      #pragma unroll
      for (int j = 0; j < 8; ++j) { cv[j] = acc[j]; ci[j] = lhalf; }
      #pragma unroll
      for (int m = 1; m <= 8; m <<= 1) {
        #pragma unroll
        for (int j = 0; j < 8; ++j) {
          float ov = __shfl_xor(cv[j], m, 32);
          int   oi = __shfl_xor(ci[j], m, 32);
          if (ov > cv[j] || (ov == cv[j] && oi < ci[j])) { cv[j] = ov; ci[j] = oi; }
        }
      }
    }

    // ---- write index maps (as float; non-temporal, write-once) -----------
    if (lhalf == 0) {
      #pragma unroll
      for (int j = 0; j < 8; ++j) {
        __builtin_nontemporal_store((float)bi[j], outS + rowBase + hi * 8 + j);
        __builtin_nontemporal_store((float)ci[j], outC + rowBase + hi * 8 + j);
      }
    }

    // broadcast winners to all lanes (rows 0..7 live in half 0, 8..15 in half 1)
    int sidx[16], cidx[16];
    #pragma unroll
    for (int r = 0; r < 16; ++r) {
      int src = (r < 8) ? 0 : 16;
      sidx[r] = __shfl(bi[r & 7], src, 32);
      cidx[r] = __shfl(ci[r & 7], src, 32);
    }

    // ---- gather quantized rows, store (NT), accumulate commitment loss ---
    float lloss = 0.0f;
    const int c = lane * 4;  // 4 cols per lane, 128 cols
    #pragma unroll
    for (int r = 0; r < 16; ++r) {
      const float* zr = in + (size_t)(rowBase + r) * D;
      const float* qsrc = (c < HALF_D)
                              ? (wS + (size_t)sidx[r] * HALF_D + c)
                              : (wC + (size_t)cidx[r] * HALF_D + (c - HALF_D));
      v4f q = *(const v4f*)qsrc;
      v4f z = *(const v4f*)(zr + c);
      float dx = q.x - z.x, dy = q.y - z.y, dz = q.z - z.z, dw = q.w - z.w;
      lloss += dx * dx + dy * dy + dz * dz + dw * dw;
      __builtin_nontemporal_store(q,
          (v4f*)(outQ + (size_t)(rowBase + r) * D + c));
    }
    #pragma unroll
    for (int m = 1; m < 32; m <<= 1) lloss += __shfl_xor(lloss, m, 32);
    if (lane == 0) lossPartial[tile] = lloss;  // one writer per tile: deterministic
  }
}

// --------------------------------------------------------------------------
// Kernel 3: deterministic fixed-order reduction of 4096 tile partials,
//           scale by COMMITMENT_COST / numel.
// --------------------------------------------------------------------------
__global__ __launch_bounds__(256) void vq_fin(const float* __restrict__ lossPartial,
                                              float* __restrict__ outLoss) {
  __shared__ float red[256];
  float s = 0.0f;
  for (int i = threadIdx.x; i < N_TILES; i += 256) s += lossPartial[i];
  red[threadIdx.x] = s;
  __syncthreads();
  #pragma unroll
  for (int off = 128; off > 0; off >>= 1) {
    if (threadIdx.x < off) red[threadIdx.x] += red[threadIdx.x + off];
    __syncthreads();
  }
  if (threadIdx.x == 0)
    *outLoss = red[0] * (0.25f / (float)((size_t)N_ROWS * D));
}

extern "C" void kernel_launch(void* const* d_in, const int* in_sizes, int n_in,
                              void* d_out, int out_size, void* d_ws,
                              size_t ws_size, hipStream_t stream) {
  const float* in = (const float*)d_in[0];   // (16,4096,128) f32
  const float* wS = (const float*)d_in[1];   // (1024,64) f32
  const float* wC = (const float*)d_in[2];   // (16,64) f32

  float* outQ    = (float*)d_out;
  float* outLoss = outQ + (size_t)N_ROWS * D;
  float* outS    = outLoss + 1;
  float* outC    = outS + N_ROWS;

  // Workspace layout: [64B pad][f16 codebook 133120B][f32 partials 16KB]
  _Float16* cb = (_Float16*)((char*)d_ws + 64);
  float* lossPartial =
      (float*)((char*)d_ws + 64 + (size_t)(KS + KC) * HALF_D * 2);

  vq_prep<<<(KS + KC) * 32 / 256, 256, 0, stream>>>(wS, wC, cb);
  vq_main<<<256, 256, 0, stream>>>(in, wS, wC, cb, outQ, outS, outC,
                                   lossPartial);
  vq_fin<<<1, 256, 0, stream>>>(lossPartial, outLoss);
}
